// gnn_atomic_lg_53712861003910
// MI455X (gfx1250) — matrix operand order repaired and checked by reading, not compiled, not run on hardware
//
#include <hip/hip_runtime.h>

// MI455X (gfx1250) wave32 WMMA implementation.
// Big gmuls: f32 -> f16 convert in LDS staging, V_WMMA_F32_16X16X32_F16 with f32 acc.
// Roofline: 6.2 GFLOP over ~200MB HBM traffic -> ~720 TFLOP/s needed to stay BW-bound;
// only f16 WMMA clears that. Software-pipelined: global_load_b128 (next tile, in flight)
// overlaps ds_load+wmma (current tile); loadcnt wait deferred to next iter's LDS stash.

typedef __attribute__((ext_vector_type(16))) _Float16     v16h;
typedef __attribute__((ext_vector_type(8)))  float        v8f;
typedef __attribute__((ext_vector_type(8)))  unsigned int v8u;

#define BB 2
#define NN 1024
#define MM 2048
#define BN_EPS 1e-5f

// ---- fragment builders (wave32 layouts per CDNA5 ISA 7.12.2) -------------
// A 16x32 f16: lanes 0-15 row=lane, VGPR v: K = (v<4 ? 2v : 16+2(v-4)); lanes 16-31: +8.
__device__ __forceinline__ v16h load_a_frag(const _Float16* lds, int base, int lid) {
  v8u au;
  const int row = lid & 15;
  const int kb  = (lid >> 4) << 3;               // 0 or 8
  const int off = base + row * 34;               // padded stride (bank-conflict free)
#pragma unroll
  for (int v = 0; v < 8; ++v) {
    const int k = kb + ((v < 4) ? (v * 2) : (16 + (v - 4) * 2));
    au[v] = *(const unsigned int*)(lds + off + k);
  }
  return __builtin_bit_cast(v16h, au);
}

// B 32x16 f16: col=lane&15, VGPR v: K = (v<4 ? 2v : 16+2(v-4)); lanes 16-31: +8.
__device__ __forceinline__ v16h load_b_frag(const _Float16* lds, int colbase, int stride, int lid) {
  v8u bu;
  const int col = colbase + (lid & 15);
  const int kb  = (lid >> 4) << 3;               // 0 or 8
  const int off = col * stride + kb;
#pragma unroll
  for (int v = 0; v < 8; ++v) {
    bu[v] = *(const unsigned int*)(lds + off + ((v < 4) ? (v * 2) : (16 + (v - 4) * 2)));
  }
  return __builtin_bit_cast(v16h, bu);
}

// ---- gmul: out[b, r, j*64+f] = sum_k W[b,r,k,j] * V[b,k,f] ---------------
// Block = 128 threads (4 waves); tile = 16 rows x (JT*64) cols; wave w owns f-tile w.
// Software pipeline: fetch(k+32) -> regs overlaps ds_load/wmma of tile k; ping-pong LDS.
template <int JT, bool TRANSP>
__global__ void __launch_bounds__(128) gmul_wmma(
    const float* __restrict__ W, const float* __restrict__ V, float* __restrict__ out,
    int Rout, int Ksrc, int sRow, int sK, int sBatchW) {
  constexpr int WE   = 16 * 32 * JT;             // W tile elements
  constexpr int WP   = WE / (4 * 128);           // float4 packs per thread (3 or 2)
  __shared__ _Float16 lds_w[2][JT * 16 * 34];
  __shared__ _Float16 lds_v[2][64 * 34];

  const int b   = blockIdx.y;
  const int r0  = blockIdx.x * 16;
  const int t   = threadIdx.x;
  const int wv  = t >> 5;                        // wave id = f-tile
  const int lid = t & 31;
  const int outC = JT * 64;

  const float* Wb = W + (long)b * sBatchW;
  const float* Vb = V + (long)b * Ksrc * 64;

  v8f zero = {};
  v8f acc[JT];
#pragma unroll
  for (int j = 0; j < JT; ++j) acc[j] = zero;

  float4 wreg[WP];
  float4 vreg[4];

  // Issue global float4 loads for K-slab k0 into registers (no wait here).
  auto fetch = [&](int k0) {
#pragma unroll
    for (int p = 0; p < WP; ++p) {
      const int flat = (t + p * 128) * 4;
      int r, k, j;
      if (TRANSP) { k = flat / (16 * JT); int rem = flat % (16 * JT); r = rem / JT; j = rem % JT; }
      else        { r = flat / (32 * JT); int rem = flat % (32 * JT); k = rem / JT; j = rem % JT; }
      const long off = (long)(r0 + r) * sRow + (long)(k0 + k) * sK + j;
      wreg[p] = *(const float4*)(Wb + off);
      if (k0 + 64 < Ksrc) __builtin_prefetch(Wb + off + 64l * sK, 0, 1); // 2 slabs ahead
    }
#pragma unroll
    for (int p = 0; p < 4; ++p) {
      const int flat = (t + p * 128) * 4;
      const int k = flat >> 6, f = flat & 63;
      vreg[p] = *(const float4*)(Vb + (long)(k0 + k) * 64 + f);
    }
  };

  // Convert fetched regs f32->f16 and scatter into LDS buffer `buf`.
  auto stash = [&](int buf) {
#pragma unroll
    for (int p = 0; p < WP; ++p) {
      const float tmp[4] = {wreg[p].x, wreg[p].y, wreg[p].z, wreg[p].w};
#pragma unroll
      for (int e = 0; e < 4; ++e) {
        const int flat = (t + p * 128) * 4 + e;
        int r, k, j;
        if (TRANSP) { k = flat / (16 * JT); int rem = flat % (16 * JT); r = rem / JT; j = rem % JT; }
        else        { r = flat / (32 * JT); int rem = flat % (32 * JT); k = rem / JT; j = rem % JT; }
        lds_w[buf][(j * 16 + r) * 34 + k] = (_Float16)tmp[e];
      }
    }
#pragma unroll
    for (int p = 0; p < 4; ++p) {
      const float tmp[4] = {vreg[p].x, vreg[p].y, vreg[p].z, vreg[p].w};
      const int flat = (t + p * 128) * 4;
      const int k = flat >> 6, f = flat & 63;
#pragma unroll
      for (int e = 0; e < 4; ++e) lds_v[buf][(f + e) * 34 + k] = (_Float16)tmp[e];
    }
  };

  fetch(0);
  int cur = 0;
  for (int k0 = 0; k0 < Ksrc; k0 += 32) {
    stash(cur);                                  // waits loadcnt, writes buf `cur`
    __syncthreads();                             // buf `cur` visible block-wide
    if (k0 + 32 < Ksrc) fetch(k0 + 32);          // next slab in flight during compute
    const v16h bfrag = load_b_frag(&lds_v[cur][0], wv * 16, 34, lid);
#pragma unroll
    for (int j = 0; j < JT; ++j) {
      const v16h afrag = load_a_frag(&lds_w[cur][0], j * 16 * 34, lid);
      acc[j] = __builtin_amdgcn_wmma_f32_16x16x32_f16(
          false, afrag, false, bfrag, (short)0, acc[j], false, false);
    }
    cur ^= 1;                                    // next stash targets other buffer
  }

  // C/D layout: VGPR v, lanes 0-15: M=v,N=lane; lanes 16-31: M=8+v,N=lane-16.
  const int coln  = lid & 15;
  const int rbase = (lid >> 4) << 3;
#pragma unroll
  for (int j = 0; j < JT; ++j) {
#pragma unroll
    for (int v = 0; v < 8; ++v) {
      out[((long)b * Rout + r0 + rbase + v) * outC + j * 64 + wv * 16 + coln] = acc[j][v];
    }
  }
}

// ---- fused dual-linear + bias + selective relu -> cat [B,R,64] -----------
// Z = [zA(192) | zB(128)], Wcat[64,320]: rows<32 from (wA1|wB1), rows>=32 from (wA2|wB2).
__global__ void __launch_bounds__(128) linear_cat_wmma(
    const float* __restrict__ zA, const float* __restrict__ zB,
    const float* __restrict__ wA1, const float* __restrict__ wB1,
    const float* __restrict__ wA2, const float* __restrict__ wB2,
    const float* __restrict__ bA1, const float* __restrict__ bB1,
    const float* __restrict__ bA2, const float* __restrict__ bB2,
    float* __restrict__ cat, int R) {
  __shared__ _Float16 lds_wt[64 * 328];          // [c][k], padded stride
  __shared__ _Float16 lds_z[16 * 34];
  const int b   = blockIdx.y;
  const int r0  = blockIdx.x * 16;
  const int t   = threadIdx.x;
  const int wv  = t >> 5;
  const int lid = t & 31;

  // One-time weight stage (64x320 f16 = 40KB LDS, reads hit L2).
  for (int i = t; i < 64 * 320; i += 128) {
    const int c = i / 320, k = i % 320;
    float v;
    if (c < 32) v = (k < 192) ? wA1[c * 192 + k] : wB1[c * 128 + (k - 192)];
    else        v = (k < 192) ? wA2[(c - 32) * 192 + k] : wB2[(c - 32) * 128 + (k - 192)];
    lds_wt[c * 328 + k] = (_Float16)v;
  }

  v8f acc = {};
  const float* zAb = zA + ((long)b * R + r0) * 192;
  const float* zBb = zB + ((long)b * R + r0) * 128;

  for (int k0 = 0; k0 < 320; k0 += 32) {
    __syncthreads();                             // covers weight stage on iter 0
    for (int i = t; i < 16 * 32; i += 128) {
      const int r = i >> 5, k = i & 31, kk = k0 + k;
      const float v = (kk < 192) ? zAb[r * 192 + kk] : zBb[r * 128 + (kk - 192)];
      lds_z[r * 34 + k] = (_Float16)v;
    }
    __syncthreads();
    const v16h a  = load_a_frag(lds_z, 0, lid);
    const v16h bw = load_b_frag(lds_wt, wv * 16, 328, lid);
    acc = __builtin_amdgcn_wmma_f32_16x16x32_f16(
        false, a, false, bw, (short)0, acc, false, false);
  }

  const int c = wv * 16 + (lid & 15);
  const float bias = (c < 32) ? (bA1[c] + bB1[c]) : (bA2[c - 32] + bB2[c - 32]);
  const int rbase = (lid >> 4) << 3;
#pragma unroll
  for (int v = 0; v < 8; ++v) {
    float val = acc[v] + bias;
    if (c < 32) val = fmaxf(val, 0.f);           // relu only on first half channels
    cat[((long)b * R + r0 + rbase + v) * 64 + c] = val;
  }
}

// ---- train-mode BN: per-channel mean / biased var over all rows ----------
__global__ void bn_stats(const float* __restrict__ catx, const float* __restrict__ caty,
                         float* __restrict__ stats) {
  const int c    = blockIdx.x;                   // 0..63
  const int side = blockIdx.y;                   // 0 = x (2048 rows), 1 = y (4096 rows)
  const float* src = side ? caty : catx;
  const int R = side ? (BB * MM) : (BB * NN);
  float s = 0.f, s2 = 0.f;
  for (int r = threadIdx.x; r < R; r += 256) {
    const float v = src[(long)r * 64 + c];
    s += v; s2 += v * v;
  }
  __shared__ float red[256], red2[256];
  red[threadIdx.x] = s; red2[threadIdx.x] = s2;
  __syncthreads();
  for (int o = 128; o > 0; o >>= 1) {
    if (threadIdx.x < o) { red[threadIdx.x] += red[threadIdx.x + o];
                           red2[threadIdx.x] += red2[threadIdx.x + o]; }
    __syncthreads();
  }
  if (threadIdx.x == 0) {
    const float m   = red[0] / R;
    const float var = red2[0] / R - m * m;
    stats[side * 128 + c]      = m;
    stats[side * 128 + 64 + c] = rsqrtf(var + BN_EPS);
  }
}

__global__ void bn_apply(const float* __restrict__ catx, const float* __restrict__ caty,
                         const float* __restrict__ stats,
                         const float* __restrict__ gx, const float* __restrict__ bx,
                         const float* __restrict__ gy, const float* __restrict__ by,
                         float* __restrict__ outx, float* __restrict__ outy) {
  const long idx = (long)blockIdx.x * blockDim.x + threadIdx.x;
  const long nx = (long)BB * NN * 64;
  const long ny = (long)BB * MM * 64;
  if (idx < nx) {
    const int c = (int)(idx & 63);
    outx[idx] = (catx[idx] - stats[c]) * stats[64 + c] * gx[c] + bx[c];
  } else if (idx < nx + ny) {
    const long i = idx - nx;
    const int c = (int)(i & 63);
    outy[i] = (caty[i] - stats[128 + c]) * stats[192 + c] * gy[c] + by[c];
  }
}

extern "C" void kernel_launch(void* const* d_in, const int* in_sizes, int n_in,
                              void* d_out, int out_size, void* d_ws, size_t ws_size,
                              hipStream_t stream) {
  (void)in_sizes; (void)n_in; (void)out_size; (void)ws_size;
  const float* WW    = (const float*)d_in[0];
  const float* x     = (const float*)d_in[1];
  const float* WW_lg = (const float*)d_in[2];
  const float* y     = (const float*)d_in[3];
  const float* P     = (const float*)d_in[4];
  const float* fcx2x_1_w = (const float*)d_in[5];
  const float* fcx2x_1_b = (const float*)d_in[6];
  const float* fcy2x_1_w = (const float*)d_in[7];
  const float* fcy2x_1_b = (const float*)d_in[8];
  const float* fcx2x_2_w = (const float*)d_in[9];
  const float* fcx2x_2_b = (const float*)d_in[10];
  const float* fcy2x_2_w = (const float*)d_in[11];
  const float* fcy2x_2_b = (const float*)d_in[12];
  const float* fcx2y_1_w = (const float*)d_in[13];
  const float* fcx2y_1_b = (const float*)d_in[14];
  const float* fcy2y_1_w = (const float*)d_in[15];
  const float* fcy2y_1_b = (const float*)d_in[16];
  const float* fcx2y_2_w = (const float*)d_in[17];
  const float* fcx2y_2_b = (const float*)d_in[18];
  const float* fcy2y_2_w = (const float*)d_in[19];
  const float* fcy2y_2_b = (const float*)d_in[20];
  const float* bn_x_g = (const float*)d_in[21];
  const float* bn_x_b = (const float*)d_in[22];
  const float* bn_y_g = (const float*)d_in[23];
  const float* bn_y_b = (const float*)d_in[24];

  // workspace layout (floats)
  float* ws   = (float*)d_ws;
  float* x2x  = ws;                                   // [B,N,192]
  float* y2x  = x2x + (long)BB * NN * 192;            // [B,N,128]
  float* y2y  = y2x + (long)BB * NN * 128;            // [B,M,192]
  float* x2y  = y2y + (long)BB * MM * 192;            // [B,M,128]
  float* catx = x2y + (long)BB * MM * 128;            // [B,N,64]
  float* caty = catx + (long)BB * NN * 64;            // [B,M,64]
  float* stats = caty + (long)BB * MM * 64;           // 256 floats

  // d_out layout: WW | x_output | WW_lg | y_output | P
  float* out      = (float*)d_out;
  const long szWW = (long)BB * NN * NN * 3;
  const long szXO = (long)BB * NN * 64;
  const long szLG = (long)BB * MM * MM * 3;
  const long szYO = (long)BB * MM * 64;
  const long szP  = (long)BB * NN * MM * 2;
  float* out_x = out + szWW;
  float* out_y = out + szWW + szXO + szLG;

  // pass-through outputs
  hipMemcpyAsync(out, WW, szWW * sizeof(float), hipMemcpyDeviceToDevice, stream);
  hipMemcpyAsync(out + szWW + szXO, WW_lg, szLG * sizeof(float), hipMemcpyDeviceToDevice, stream);
  hipMemcpyAsync(out + szWW + szXO + szLG + szYO, P, szP * sizeof(float), hipMemcpyDeviceToDevice, stream);

  // gmuls (WMMA f16, software-pipelined)
  gmul_wmma<3, false><<<dim3(NN / 16, BB), 128, 0, stream>>>(WW,    x, x2x, NN, NN, NN * 3, 3, NN * NN * 3);
  gmul_wmma<2, false><<<dim3(NN / 16, BB), 128, 0, stream>>>(P,     y, y2x, NN, MM, MM * 2, 2, NN * MM * 2);
  gmul_wmma<3, false><<<dim3(MM / 16, BB), 128, 0, stream>>>(WW_lg, y, y2y, MM, MM, MM * 3, 3, MM * MM * 3);
  gmul_wmma<2, true ><<<dim3(MM / 16, BB), 128, 0, stream>>>(P,     x, x2y, MM, NN, 2, MM * 2, NN * MM * 2);

  // fused linears + relu/concat
  linear_cat_wmma<<<dim3(NN / 16, BB), 128, 0, stream>>>(
      x2x, y2x, fcx2x_1_w, fcy2x_1_w, fcx2x_2_w, fcy2x_2_w,
      fcx2x_1_b, fcy2x_1_b, fcx2x_2_b, fcy2x_2_b, catx, NN);
  linear_cat_wmma<<<dim3(MM / 16, BB), 128, 0, stream>>>(
      y2y, x2y, fcy2y_1_w, fcx2y_1_w, fcy2y_2_w, fcx2y_2_w,
      fcy2y_1_b, fcx2y_1_b, fcy2y_2_b, fcx2y_2_b, caty, MM);

  // batchnorm (train-mode stats) + write outputs
  bn_stats<<<dim3(64, 2), 256, 0, stream>>>(catx, caty, stats);
  const long totalBN = (long)BB * NN * 64 + (long)BB * MM * 64;
  bn_apply<<<(unsigned)((totalBN + 255) / 256), 256, 0, stream>>>(
      catx, caty, stats, bn_x_g, bn_x_b, bn_y_g, bn_y_b, out_x, out_y);
}
